// Block_74964359184941
// MI455X (gfx1250) — matrix operand order repaired and checked by reading, not compiled, not run on hardware
//
#include <hip/hip_runtime.h>
#include <hip/hip_bf16.h>

// ---- problem constants --------------------------------------------------
#define BB   4
#define TT   2048
#define CC   1024
#define HH   16
#define HD   64
#define MTOT (BB*TT)      // 8192 rows
#define FF   (4*CC)       // 4096

typedef __attribute__((ext_vector_type(16))) __bf16 bf16x16;
typedef __attribute__((ext_vector_type(8)))  float  floatx8;
typedef __attribute__((ext_vector_type(4)))  unsigned uint32x4;
typedef __attribute__((ext_vector_type(4)))  int      int32x4;
typedef __attribute__((ext_vector_type(8)))  int      int32x8;

// ---- helpers ------------------------------------------------------------
__device__ __forceinline__ void sched_fence(){
#if __has_builtin(__builtin_amdgcn_sched_barrier)
  __builtin_amdgcn_sched_barrier(0);   // keep loads batched before the WMMA chain
#endif
}

__device__ __forceinline__ unsigned short f2bf(float f){
  union { float f; unsigned u; } v; v.f = f;
  unsigned r = v.u + 0x7fffu + ((v.u >> 16) & 1u);   // round-to-nearest-even
  return (unsigned short)(r >> 16);
}

__device__ __forceinline__ floatx8 zero8(){
  floatx8 z = {0.f,0.f,0.f,0.f,0.f,0.f,0.f,0.f};
  return z;
}

__device__ __forceinline__ floatx8 wmma_bf16(bf16x16 a, bf16x16 b, floatx8 c){
  return __builtin_amdgcn_wmma_f32_16x16x32_bf16(false, a, false, b, (short)0, c,
                                                 false, false);
}

// A fragment: 16x32 bf16 tile, row-major with stride lda.
// ISA 7.12.2 16-bit A layout: lane m=L&15, half=L>>4;
// vgpr j holds k = (j&3)*2 + (j>>2)*16 + 8*half (packed pair).
__device__ __forceinline__ bf16x16 load_a_frag(const unsigned short* A, int lda, int lane){
  const int m = lane & 15, hf = lane >> 4;
  union { bf16x16 v; unsigned u[8]; } f;
#pragma unroll
  for (int j = 0; j < 8; ++j){
    const int k = ((j & 3) << 1) + ((j >> 2) << 4) + (hf << 3);
    f.u[j] = *(const unsigned*)(A + m*lda + k);
  }
  return f.v;
}

// B fragment: 32x16 bf16 (KxN), given as Bt[n][k] row-major, stride ldb.
// same k order as A: vgpr j holds k = (j&3)*2 + (j>>2)*16 + 8*half (packed pair).
__device__ __forceinline__ bf16x16 load_b_frag(const unsigned short* Bt, int ldb, int lane){
  const int n = lane & 15, hf = lane >> 4;
  union { bf16x16 v; unsigned u[8]; } f;
#pragma unroll
  for (int j = 0; j < 8; ++j){
    const int k = ((j & 3) << 1) + ((j >> 2) << 4) + (hf << 3);
    f.u[j] = *(const unsigned*)(Bt + n*ldb + k);
  }
  return f.v;
}

// ---- Tensor Data Mover: load a 64(n) x 32(k) bf16 panel into LDS --------
// D# per CDNA5 ISA. Tile is always fully in bounds:
// tensor dims == tile dims, dim0 stride = ldb elements. LDS dest is packed
// [64][32] bf16 (row = 64 bytes), which is exactly the layout load_b_frag
// expects with ldb_lds = 32.
__device__ __forceinline__ void tdm_load_b_panel(const unsigned short* gsrc,
                                                 unsigned lds_off, int ldb){
#if __has_builtin(__builtin_amdgcn_tensor_load_to_lds)
  const unsigned long long ga = (unsigned long long)(uintptr_t)gsrc;
  uint32x4 g0;
  g0[0] = 1u;                                           // count=1, user mode
  g0[1] = lds_off;                                      // lds_addr (bytes)
  g0[2] = (unsigned)ga;                                 // global_addr[31:0]
  g0[3] = (unsigned)((ga >> 32) & 0x01ffffffu) | (2u << 30);  // addr[56:32] | type=2
  int32x8 g1;
  g1[0] = (int)(1u << 16);      // workgroup_mask=0, data_size=1 (2 bytes)
  g1[1] = (int)(32u << 16);     // tensor_dim0[15:0]=32 at bits 63:48
  g1[2] = (int)(64u << 16);     // tensor_dim0 hi=0, tensor_dim1[15:0]=64
  g1[3] = (int)(32u << 16);     // tensor_dim1 hi=0, tile_dim0=32
  g1[4] = (int)(64u);           // tile_dim1=64, tile_dim2=0
  g1[5] = ldb;                  // tensor_dim0_stride[31:0] (elements)
  g1[6] = 0;                    // stride hi, tensor_dim1_stride lo
  g1[7] = 0;
  int32x4 z4 = {0,0,0,0};
#if __clang_major__ >= 23
  int32x8 z8 = {0,0,0,0,0,0,0,0};
  __builtin_amdgcn_tensor_load_to_lds(g0, g1, z4, z4, z8, 0);
#else
  __builtin_amdgcn_tensor_load_to_lds(g0, g1, z4, z4, 0);
#endif
#else
  (void)gsrc; (void)lds_off; (void)ldb;
#endif
}

__device__ __forceinline__ void tdm_wait(){
#if __has_builtin(__builtin_amdgcn_s_wait_tensorcnt)
  __builtin_amdgcn_s_wait_tensorcnt(0);
#else
  asm volatile("s_wait_tensorcnt 0x0" ::: "memory");
#endif
}

// ---- shared-B GEMM: block = 8 waves, 128(M) x 64(N) tile ----------------
// B panel (64 x 32 bf16 / k-step) staged in LDS by the TDM, double buffered,
// shared by all 8 waves. A fragments stream from global per wave. All four
// B fragments are loaded BEFORE a sched_barrier so the DS loads batch into
// one clause, get distinct registers, and the 4 WMMAs issue back-to-back
// behind a single s_wait.
#define BPANEL_ELTS (64*32)

__device__ __forceinline__ void gemm_16x64_ldsB(const unsigned short* __restrict__ A, int lda,
                                                const unsigned short* __restrict__ Bt, int ldb,
                                                int K, floatx8 acc[4], int lane, int wave,
                                                unsigned short* bpanel /* [2][64*32] */){
#if __has_builtin(__builtin_amdgcn_tensor_load_to_lds)
  if (wave == 0){
    tdm_load_b_panel(Bt, (unsigned)(uintptr_t)bpanel, ldb);
    tdm_wait();
  }
  __syncthreads();
  for (int k0 = 0; k0 < K; k0 += 32){
    const int cur = (k0 >> 5) & 1;
    if (wave == 0 && (k0 + 32) < K)
      tdm_load_b_panel(Bt + k0 + 32,
                       (unsigned)(uintptr_t)(bpanel + (1-cur)*BPANEL_ELTS), ldb);
    __builtin_prefetch(A + k0 + 128, 0, 0);
    bf16x16 a = load_a_frag(A + k0, lda, lane);
    const unsigned short* bp = bpanel + cur*BPANEL_ELTS;
    bf16x16 b0 = load_b_frag(bp + 0*16*32, 32, lane);
    bf16x16 b1 = load_b_frag(bp + 1*16*32, 32, lane);
    bf16x16 b2 = load_b_frag(bp + 2*16*32, 32, lane);
    bf16x16 b3 = load_b_frag(bp + 3*16*32, 32, lane);
    sched_fence();
    acc[0] = wmma_bf16(a, b0, acc[0]);
    acc[1] = wmma_bf16(a, b1, acc[1]);
    acc[2] = wmma_bf16(a, b2, acc[2]);
    acc[3] = wmma_bf16(a, b3, acc[3]);
    if (wave == 0) tdm_wait();
    __syncthreads();
  }
#else
  // fallback: direct global B fragments (still WMMA)
  for (int k0 = 0; k0 < K; k0 += 32){
    __builtin_prefetch(A + k0 + 128, 0, 0);
    bf16x16 a = load_a_frag(A + k0, lda, lane);
    bf16x16 b0 = load_b_frag(Bt + (size_t)( 0)*ldb + k0, ldb, lane);
    bf16x16 b1 = load_b_frag(Bt + (size_t)(16)*ldb + k0, ldb, lane);
    bf16x16 b2 = load_b_frag(Bt + (size_t)(32)*ldb + k0, ldb, lane);
    bf16x16 b3 = load_b_frag(Bt + (size_t)(48)*ldb + k0, ldb, lane);
    sched_fence();
    acc[0] = wmma_bf16(a, b0, acc[0]);
    acc[1] = wmma_bf16(a, b1, acc[1]);
    acc[2] = wmma_bf16(a, b2, acc[2]);
    acc[3] = wmma_bf16(a, b3, acc[3]);
  }
  (void)wave; (void)bpanel;
#endif
}

// ---- weight conversion --------------------------------------------------
// wqkvT[sel][h][d][c] = w_sel[h][c][d]  (bf16, B-transposed layout)
__global__ void conv_qkv_kernel(const float* __restrict__ wq, const float* __restrict__ wk,
                                const float* __restrict__ wv, unsigned short* __restrict__ wqkvT){
  size_t i = (size_t)blockIdx.x*256 + threadIdx.x;
  if (i >= (size_t)3*HH*HD*CC) return;
  int c = (int)(i & (CC-1)); size_t j = i >> 10;
  int d = (int)(j & (HD-1)); j >>= 6;
  int h = (int)(j & (HH-1)); int sel = (int)(j >> 4);
  const float* w = (sel == 0) ? wq : (sel == 1) ? wk : wv;
  wqkvT[i] = f2bf(w[((size_t)h*CC + c)*HD + d]);
}

// wT[n][k] = w[k][n]
__global__ void conv_T_kernel(const float* __restrict__ w, unsigned short* __restrict__ wT,
                              int K, int N){
  size_t i = (size_t)blockIdx.x*256 + threadIdx.x;
  if (i >= (size_t)K*N) return;
  int k = (int)(i % (size_t)K);
  int n = (int)(i / (size_t)K);
  wT[i] = f2bf(w[(size_t)k*N + n]);
}

// ---- layer norm (fp32 in -> bf16 out) -----------------------------------
__global__ void ln_kernel(const float* __restrict__ x, const float* __restrict__ g,
                          const float* __restrict__ bta, unsigned short* __restrict__ out){
  __shared__ float s1[256], s2[256];
  const int row = blockIdx.x, tid = threadIdx.x;
  const float* xr = x + (size_t)row*CC;
  float v[4], s = 0.f, ss = 0.f;
#pragma unroll
  for (int i = 0; i < 4; ++i){ v[i] = xr[tid + i*256]; s += v[i]; ss += v[i]*v[i]; }
  s1[tid] = s; s2[tid] = ss; __syncthreads();
  for (int o = 128; o > 0; o >>= 1){
    if (tid < o){ s1[tid] += s1[tid+o]; s2[tid] += s2[tid+o]; }
    __syncthreads();
  }
  const float mean = s1[0] * (1.f/CC);
  const float var  = s2[0] * (1.f/CC) - mean*mean;
  const float rstd = rsqrtf(var + 1e-5f);
#pragma unroll
  for (int i = 0; i < 4; ++i){
    const int c = tid + i*256;
    out[(size_t)row*CC + c] = f2bf((v[i]-mean)*rstd*g[c] + bta[c]);
  }
}

// ---- QKV projection -----------------------------------------------------
// grid.x = M/128, grid.y = 3*H (one 64-wide head slice per y)
__global__ void qkv_kernel(const unsigned short* __restrict__ hb,
                           const unsigned short* __restrict__ wqkvT,
                           unsigned short* __restrict__ qb,
                           unsigned short* __restrict__ kb,
                           unsigned short* __restrict__ vTb){
  __shared__ __align__(16) unsigned short bpanel[2*BPANEL_ELTS];
  const int lane = threadIdx.x & 31, wave = threadIdx.x >> 5;
  const int m0 = blockIdx.x*128 + wave*16;
  const int ntile = blockIdx.y;          // 0..47
  const int sel = ntile >> 4, hh = ntile & 15;
  floatx8 acc[4] = {zero8(), zero8(), zero8(), zero8()};
  gemm_16x64_ldsB(hb + (size_t)m0*CC, CC, wqkvT + (size_t)ntile*HD*CC, CC, CC,
                  acc, lane, wave, bpanel);
  const int n = lane & 15, hf = lane >> 4;
#pragma unroll
  for (int nn = 0; nn < 4; ++nn){
#pragma unroll
    for (int r = 0; r < 8; ++r){
      const int gm = m0 + r + 8*hf;
      const int b = gm >> 11, t = gm & (TT-1);
      const int d = nn*16 + n;
      const unsigned short val = f2bf(acc[nn][r]);
      if (sel == 0)      qb [(((size_t)b*HH + hh)*TT + t)*HD + d] = val;
      else if (sel == 1) kb [(((size_t)b*HH + hh)*TT + t)*HD + d] = val;
      else               vTb[(((size_t)b*HH + hh)*HD + d)*TT + t] = val;   // V stored transposed
    }
  }
}

// ---- causal flash attention (one 16-row Q tile per wave) ---------------
__global__ void attn_kernel(const unsigned short* __restrict__ qb,
                            const unsigned short* __restrict__ kb,
                            const unsigned short* __restrict__ vTb,
                            unsigned short* __restrict__ attb){
  __shared__ __align__(16) unsigned short lds_p[8][16*32];   // per-wave P staging
  const int lane = threadIdx.x & 31, wave = threadIdx.x >> 5;
  const int wid = blockIdx.x*8 + wave;              // 0..8191
  const int qt = wid & 127, bh = wid >> 7;          // bh = b*16+h
  const int b = bh >> 4, hh = bh & 15;
  const int r0 = qt*16;
  const int n = lane & 15, hf = lane >> 4;

  const unsigned short* qp = qb  + ((size_t)bh*TT + r0)*HD;
  const unsigned short* kp = kb  + (size_t)bh*TT*HD;
  const unsigned short* vp = vTb + (size_t)bh*HD*TT;

  const bf16x16 aq0 = load_a_frag(qp +  0, HD, lane);
  const bf16x16 aq1 = load_a_frag(qp + 32, HD, lane);

  float mrow[8], lrow[8];
  floatx8 o[4] = {zero8(), zero8(), zero8(), zero8()};
#pragma unroll
  for (int r = 0; r < 8; ++r){ mrow[r] = -1.0e30f; lrow[r] = 0.f; }

  unsigned short* lp = &lds_p[wave][0];

  for (int s0 = 0; s0 < r0 + 16; s0 += 32){
    // S(16x32) = Q(16x64) * K^T chunk, two 16-wide column groups.
    // Load all four K fragments first, then issue the WMMA chain.
    bf16x16 bk00 = load_b_frag(kp + (size_t)(s0     )*HD +  0, HD, lane);
    bf16x16 bk01 = load_b_frag(kp + (size_t)(s0     )*HD + 32, HD, lane);
    bf16x16 bk10 = load_b_frag(kp + (size_t)(s0 + 16)*HD +  0, HD, lane);
    bf16x16 bk11 = load_b_frag(kp + (size_t)(s0 + 16)*HD + 32, HD, lane);
    sched_fence();
    floatx8 sg[2];
    sg[0] = wmma_bf16(aq1, bk01, wmma_bf16(aq0, bk00, zero8()));
    sg[1] = wmma_bf16(aq1, bk11, wmma_bf16(aq0, bk10, zero8()));
    // scale + causal mask
#pragma unroll
    for (int r = 0; r < 8; ++r){
      const int t = r0 + r + 8*hf;
#pragma unroll
      for (int g = 0; g < 2; ++g){
        const int s = s0 + g*16 + n;
        const float v = sg[g][r] * 0.125f;          // 1/sqrt(64)
        sg[g][r] = (s <= t) ? v : -3.0e38f;
      }
    }
    // online softmax (row reductions across 16 lanes of each half)
#pragma unroll
    for (int r = 0; r < 8; ++r){
      float mx = fmaxf(sg[0][r], sg[1][r]);
#pragma unroll
      for (int off = 1; off < 16; off <<= 1) mx = fmaxf(mx, __shfl_xor(mx, off, 32));
      const float mnew = fmaxf(mrow[r], mx);
      const float corr = __expf(mrow[r] - mnew);
      const float p0 = __expf(sg[0][r] - mnew);
      const float p1 = __expf(sg[1][r] - mnew);
      float rs = p0 + p1;
#pragma unroll
      for (int off = 1; off < 16; off <<= 1) rs += __shfl_xor(rs, off, 32);
      lrow[r] = lrow[r]*corr + rs;
      mrow[r] = mnew;
#pragma unroll
      for (int nn = 0; nn < 4; ++nn) o[nn][r] = o[nn][r]*corr;
      sg[0][r] = p0; sg[1][r] = p1;
    }
    // P: C-layout -> A-fragment layout via per-wave LDS slice
#pragma unroll
    for (int r = 0; r < 8; ++r){
#pragma unroll
      for (int g = 0; g < 2; ++g)
        lp[(r + 8*hf)*32 + g*16 + n] = f2bf(sg[g][r]);
    }
    asm volatile("s_wait_dscnt 0" ::: "memory");
    const bf16x16 pa = load_a_frag(lp, 32, lane);
    // O(16x64) += P(16x32) * V(32x64)   (V^T rows are d, contiguous in s)
    bf16x16 bv0 = load_b_frag(vp + (size_t)( 0)*TT + s0, TT, lane);
    bf16x16 bv1 = load_b_frag(vp + (size_t)(16)*TT + s0, TT, lane);
    bf16x16 bv2 = load_b_frag(vp + (size_t)(32)*TT + s0, TT, lane);
    bf16x16 bv3 = load_b_frag(vp + (size_t)(48)*TT + s0, TT, lane);
    sched_fence();
    o[0] = wmma_bf16(pa, bv0, o[0]);
    o[1] = wmma_bf16(pa, bv1, o[1]);
    o[2] = wmma_bf16(pa, bv2, o[2]);
    o[3] = wmma_bf16(pa, bv3, o[3]);
  }

  // normalize and emit att as [B,T,C] bf16 (heads concatenated)
#pragma unroll
  for (int r = 0; r < 8; ++r){
    const float inv = 1.f / lrow[r];
    const int t = r0 + r + 8*hf;
#pragma unroll
    for (int nn = 0; nn < 4; ++nn){
      const int d = nn*16 + n;
      attb[((size_t)b*TT + t)*CC + hh*HD + d] = f2bf(o[nn][r]*inv);
    }
  }
}

// ---- output projection + residual ---------------------------------------
__global__ void proj_kernel(const unsigned short* __restrict__ attb,
                            const unsigned short* __restrict__ wpT,
                            const float* __restrict__ x,
                            const float* __restrict__ bproj,
                            float* __restrict__ x1){
  __shared__ __align__(16) unsigned short bpanel[2*BPANEL_ELTS];
  const int lane = threadIdx.x & 31, wave = threadIdx.x >> 5;
  const int m0 = blockIdx.x*128 + wave*16;
  const int n0 = blockIdx.y*64;
  floatx8 acc[4] = {zero8(), zero8(), zero8(), zero8()};
  gemm_16x64_ldsB(attb + (size_t)m0*CC, CC, wpT + (size_t)n0*CC, CC, CC,
                  acc, lane, wave, bpanel);
  const int n = lane & 15, hf = lane >> 4;
#pragma unroll
  for (int nn = 0; nn < 4; ++nn){
#pragma unroll
    for (int r = 0; r < 8; ++r){
      const int gm = m0 + r + 8*hf, col = n0 + nn*16 + n;
      const size_t idx = (size_t)gm*CC + col;
      x1[idx] = x[idx] + acc[nn][r] + bproj[col];
    }
  }
}

// ---- FFN layer 1: relu(h2 @ w1 + b1) -> bf16 -----------------------------
__global__ void ff1_kernel(const unsigned short* __restrict__ h2,
                           const unsigned short* __restrict__ w1T,
                           const float* __restrict__ b1,
                           unsigned short* __restrict__ y1){
  __shared__ __align__(16) unsigned short bpanel[2*BPANEL_ELTS];
  const int lane = threadIdx.x & 31, wave = threadIdx.x >> 5;
  const int m0 = blockIdx.x*128 + wave*16;
  const int n0 = blockIdx.y*64;
  floatx8 acc[4] = {zero8(), zero8(), zero8(), zero8()};
  gemm_16x64_ldsB(h2 + (size_t)m0*CC, CC, w1T + (size_t)n0*CC, CC, CC,
                  acc, lane, wave, bpanel);
  const int n = lane & 15, hf = lane >> 4;
#pragma unroll
  for (int nn = 0; nn < 4; ++nn){
#pragma unroll
    for (int r = 0; r < 8; ++r){
      const int gm = m0 + r + 8*hf, col = n0 + nn*16 + n;
      y1[(size_t)gm*FF + col] = f2bf(fmaxf(acc[nn][r] + b1[col], 0.f));
    }
  }
}

// ---- FFN layer 2 + residual -> final output ------------------------------
__global__ void ff2_kernel(const unsigned short* __restrict__ y1,
                           const unsigned short* __restrict__ w2T,
                           const float* __restrict__ b2,
                           const float* __restrict__ x1,
                           float* __restrict__ out){
  __shared__ __align__(16) unsigned short bpanel[2*BPANEL_ELTS];
  const int lane = threadIdx.x & 31, wave = threadIdx.x >> 5;
  const int m0 = blockIdx.x*128 + wave*16;
  const int n0 = blockIdx.y*64;
  floatx8 acc[4] = {zero8(), zero8(), zero8(), zero8()};
  gemm_16x64_ldsB(y1 + (size_t)m0*FF, FF, w2T + (size_t)n0*FF, FF, FF,
                  acc, lane, wave, bpanel);
  const int n = lane & 15, hf = lane >> 4;
#pragma unroll
  for (int nn = 0; nn < 4; ++nn){
#pragma unroll
    for (int r = 0; r < 8; ++r){
      const int gm = m0 + r + 8*hf, col = n0 + nn*16 + n;
      const size_t idx = (size_t)gm*CC + col;
      out[idx] = x1[idx] + acc[nn][r] + b2[col];
    }
  }
}

// ---- host-side orchestration --------------------------------------------
extern "C" void kernel_launch(void* const* d_in, const int* in_sizes, int n_in,
                              void* d_out, int out_size, void* d_ws, size_t ws_size,
                              hipStream_t stream){
  (void)in_sizes; (void)n_in; (void)out_size; (void)ws_size;
  const float* x      = (const float*)d_in[0];
  const float* wq     = (const float*)d_in[1];
  const float* wk     = (const float*)d_in[2];
  const float* wv     = (const float*)d_in[3];
  const float* w_proj = (const float*)d_in[4];
  const float* b_proj = (const float*)d_in[5];
  const float* w1     = (const float*)d_in[6];
  const float* b1     = (const float*)d_in[7];
  const float* w2     = (const float*)d_in[8];
  const float* b2     = (const float*)d_in[9];
  const float* ln1_g  = (const float*)d_in[10];
  const float* ln1_b  = (const float*)d_in[11];
  const float* ln2_g  = (const float*)d_in[12];
  const float* ln2_b  = (const float*)d_in[13];
  float* out = (float*)d_out;

  char* ws = (char*)d_ws;
  size_t off = 0;
  auto alloc = [&](size_t bytes)->char*{
    off = (off + 255) & ~(size_t)255;
    char* p = ws + off; off += bytes; return p;
  };
  unsigned short* wqkvT = (unsigned short*)alloc((size_t)3*CC*CC*2);
  unsigned short* wpT   = (unsigned short*)alloc((size_t)CC*CC*2);
  unsigned short* w1T   = (unsigned short*)alloc((size_t)CC*FF*2);
  unsigned short* w2T   = (unsigned short*)alloc((size_t)FF*CC*2);
  unsigned short* hbuf  = (unsigned short*)alloc((size_t)MTOT*CC*2);
  unsigned short* qb    = (unsigned short*)alloc((size_t)MTOT*CC*2);
  unsigned short* kb    = (unsigned short*)alloc((size_t)MTOT*CC*2);
  unsigned short* vTb   = (unsigned short*)alloc((size_t)MTOT*CC*2);
  unsigned short* attb  = (unsigned short*)alloc((size_t)MTOT*CC*2);
  float*          x1b   = (float*)         alloc((size_t)MTOT*CC*4);
  unsigned short* h2b   = (unsigned short*)alloc((size_t)MTOT*CC*2);
  unsigned short* y1b   = (unsigned short*)alloc((size_t)MTOT*FF*2);

  // weight conversions (bf16, B-transposed layouts)
  conv_qkv_kernel<<<(3*CC*CC + 255)/256, 256, 0, stream>>>(wq, wk, wv, wqkvT);
  conv_T_kernel<<<(CC*CC + 255)/256, 256, 0, stream>>>(w_proj, wpT, CC, CC);
  conv_T_kernel<<<(CC*FF + 255)/256, 256, 0, stream>>>(w1, w1T, CC, FF);
  conv_T_kernel<<<(FF*CC + 255)/256, 256, 0, stream>>>(w2, w2T, FF, CC);

  // LN1 -> h (bf16)
  ln_kernel<<<MTOT, 256, 0, stream>>>(x, ln1_g, ln1_b, hbuf);
  // QKV projections
  qkv_kernel<<<dim3(MTOT/128, 3*HH), 256, 0, stream>>>(hbuf, wqkvT, qb, kb, vTb);
  // causal attention
  attn_kernel<<<(BB*HH*(TT/16))/8, 256, 0, stream>>>(qb, kb, vTb, attb);
  // out-proj + residual -> x1 (fp32)
  proj_kernel<<<dim3(MTOT/128, CC/64), 256, 0, stream>>>(attb, wpT, x, b_proj, x1b);
  // LN2 -> h2 (bf16)
  ln_kernel<<<MTOT, 256, 0, stream>>>(x1b, ln2_g, ln2_b, h2b);
  // FFN
  ff1_kernel<<<dim3(MTOT/128, FF/64), 256, 0, stream>>>(h2b, w1T, b1, y1b);
  ff2_kernel<<<dim3(MTOT/128, CC/64), 256, 0, stream>>>(y1b, w2T, b2, x1b, out);
}
